// SlotClassifier_7129645711763
// MI455X (gfx1250) — matrix operand order repaired and checked by reading, not compiled, not run on hardware
//
#include <hip/hip_runtime.h>
#include <hip/hip_bf16.h>

// ---------------------------------------------------------------------------
// SlotClassifier forward, MI455X (gfx1250, wave32, WMMA).
// One block = one image; all activations in LDS (325,632 B peak).
// All GEMMs via v_wmma_f32_16x16x32_f16 with pre-packed / contiguous fragments.
// Branchless transcendentals (A&S erf + v_exp_f32) for the VALU tail.
// ---------------------------------------------------------------------------

typedef __attribute__((ext_vector_type(16))) _Float16 v16h;
typedef __attribute__((ext_vector_type(8)))  _Float16 v8h;
typedef __attribute__((ext_vector_type(8)))  float    v8f;

#define NTOK 196
#define NPAD 208   // 13 * 16
#define DMODEL 64

// f16 weight workspace layout (element offsets; all multiples of 16)
#define WS_CONVW 0          // 3*10*512 = 15360 packed B-fragments
#define WS_PROJW 15360      // 5*4*512  = 10240
#define WS_WQKV  25600      // 2*12*512 = 12288
#define WS_WO    37888      // 2*4*512  = 4096
#define WS_W1    41984      // 2*16*512 = 16384
#define WS_W2    58368      // 8*4*512  = 16384
#define WS_SLOTA 74752      // 2*512    = 1024 packed A-fragments (l2n slot_q)

// LDS arena (bytes)
#define LDS_TOK   0         // 208*64 f32 = 53248 (tokens, persistent)
#define LDS_A16   53248     // 208*64 f16 = 26624 (LN out / attn-O / l2n tokens)
#define LDS_BIG   79872     // phase-overlaid big arena (245760 B)
#define LDS_TOTAL 325632

__device__ __forceinline__ v8f wmma_f16(v16h a, v16h b, v8f c) {
  return __builtin_amdgcn_wmma_f32_16x16x32_f16(false, a, false, b, (short)0, c,
                                                false, false);
}

// Branchless erf (Abramowitz-Stegun 7.1.26, |err| < 1.5e-7)
__device__ __forceinline__ float erf_fast(float x) {
  float s = __builtin_copysignf(1.0f, x);
  float a = __builtin_fabsf(x);
  float t = __builtin_amdgcn_rcpf(1.0f + 0.3275911f * a);
  float y = t * (0.254829592f +
            t * (-0.284496736f +
            t * (1.421413741f +
            t * (-1.453152027f +
            t * 1.061405429f))));
  return s * (1.0f - y * __expf(-a * a));
}

__device__ __forceinline__ float gelu_exact(float x) {
  return 0.5f * x * (1.0f + erf_fast(x * 0.70710678118654752f));
}

// A fragment (16x32 f16) from row-major [M x lda], element e of lane:
//   K = k0 + e + (lane>=16 ? 16 : 0)
// One contiguous 32B run per lane. FULL=false: K-chunk only 16 valid (hi-lanes 0).
template <bool FULL>
__device__ __forceinline__ v16h load_a_frag(const _Float16* A, int lda, int m0,
                                            int k0, int lane) {
  const int l = lane & 15, hi = lane >> 4;
  const _Float16* p = A + (m0 + l) * lda + k0 + hi * 16;
  v8h a0 = {};
  v8h a1 = {};
  if (FULL || !hi) { a0 = *(const v8h*)p; a1 = *(const v8h*)(p + 8); }
  v16h f;
#pragma unroll
  for (int e = 0; e < 8; ++e) { f[e] = a0[e]; f[e + 8] = a1[e]; }
  return f;
}

// B fragment (32x16 f16) from "n-major" storage S[n*ldb + k] (pairs along K
// are contiguous). Element e of lane: K = k0 + e + (lane>=16 ? 16 : 0),
// N = n0 + (lane&15). One contiguous 32B run. FULL=false: hi-lanes are zero.
template <bool FULL>
__device__ __forceinline__ v16h load_b_frag(const _Float16* S, int ldb, int k0,
                                            int n0, int lane) {
  const int n = n0 + (lane & 15), hi = lane >> 4;
  const _Float16* p = S + n * ldb + k0;
  if (FULL) return *(const v16h*)(p + hi * 16);
  if (hi) { v16h z = {}; return z; }
  return *(const v16h*)p;
}

// Pre-packed weight B fragment: single aligned 32B load.
__device__ __forceinline__ v16h load_wfrag(const _Float16* P, int kc, int NT,
                                           int nt, int lane) {
  return *(const v16h*)(P + (((kc * NT) + nt) * 32 + lane) * 16);
}

__device__ __forceinline__ void ln_rows_to_f16(const float* tok, _Float16* dst,
                                               const float* g, const float* bta,
                                               int tid) {
  for (int row = tid; row < NPAD; row += 256) {
    _Float16* o = dst + row * DMODEL;
    if (row < NTOK) {
      const float* t = tok + row * DMODEL;
      float mu = 0.f;
      for (int c = 0; c < DMODEL; ++c) mu += t[c];
      mu *= (1.0f / DMODEL);
      float var = 0.f;
      for (int c = 0; c < DMODEL; ++c) { float d = t[c] - mu; var += d * d; }
      var *= (1.0f / DMODEL);
      float r = rsqrtf(var + 1e-5f);
      for (int c = 0; c < DMODEL; ++c)
        o[c] = (_Float16)((t[c] - mu) * r * g[c] + bta[c]);
    } else {
      for (int c = 0; c < DMODEL; ++c) o[c] = (_Float16)0.0f;
    }
  }
}

// ---------------------------------------------------------------------------
// Prep: pack weights f32 -> per-lane f16 B-fragment images; l2n slot queries.
// Pack index: P[(kc*NT+nt)*512 + lane*16 + e] = W[n, k] with
//   n = nt*16 + (lane&15), k = kc*32 + (lane>=16?16:0) + e (0 if OOB).
// ---------------------------------------------------------------------------
__device__ void pack_bt(const float* __restrict__ W, int N, int K, int KC, int NT,
                        _Float16* __restrict__ dst, int tid) {
  const int total = KC * NT * 512;
  for (int idx = tid; idx < total; idx += 256) {
    int e = idx & 15, lane = (idx >> 4) & 31, t = idx >> 9;
    int kc = t / NT, nt = t - kc * NT;
    int n = nt * 16 + (lane & 15);
    int k = kc * 32 + ((lane >> 4) ? 16 : 0) + e;
    float v = (n < N && k < K) ? W[n * K + k] : 0.f;
    dst[idx] = (_Float16)v;
  }
}

__global__ void slotnet_prep(const float* __restrict__ convw,
                             const float* __restrict__ projw,
                             const float* __restrict__ wqkv,
                             const float* __restrict__ wo,
                             const float* __restrict__ w1,
                             const float* __restrict__ w2,
                             const float* __restrict__ slotq,
                             _Float16* __restrict__ ws) {
  __shared__ float qn[12 * 64];
  const int tid = threadIdx.x;
  pack_bt(convw, 150,  81, 3, 10, ws + WS_CONVW, tid);
  pack_bt(projw,  64, 150, 5,  4, ws + WS_PROJW, tid);
  pack_bt(wqkv,  192,  64, 2, 12, ws + WS_WQKV,  tid);
  pack_bt(wo,     64,  64, 2,  4, ws + WS_WO,    tid);
  pack_bt(w1,    256,  64, 2, 16, ws + WS_W1,    tid);
  pack_bt(w2,     64, 256, 8,  4, ws + WS_W2,    tid);
  if (tid < 12) {
    float n2 = 0.f;
    for (int c = 0; c < 64; ++c) { float v = slotq[tid * 64 + c]; n2 += v * v; }
    float inv = 1.0f / fmaxf(sqrtf(n2), 1e-12f);
    for (int c = 0; c < 64; ++c) qn[tid * 64 + c] = slotq[tid * 64 + c] * inv;
  }
  __syncthreads();
  // pack slot-query A fragments (A-frag element mapping)
  for (int idx = tid; idx < 2 * 512; idx += 256) {
    int e = idx & 15, lane = (idx >> 4) & 31, kc = idx >> 9;
    int m = lane & 15, hi = lane >> 4;
    int k = kc * 32 + e + hi * 16;
    float v = (m < 12 && k < 64) ? qn[m * 64 + k] : 0.f;
    ws[WS_SLOTA + idx] = (_Float16)v;
  }
}

// ---------------------------------------------------------------------------
// Megakernel: one image per block, 256 threads = 8 wave32 waves.
// ---------------------------------------------------------------------------
__global__ __launch_bounds__(256, 1) void slotnet_main(
    const float* __restrict__ x,
    const float* __restrict__ proj_b,
    const float* __restrict__ ln1_g, const float* __restrict__ ln1_b,
    const float* __restrict__ bqkv,  const float* __restrict__ bo,
    const float* __restrict__ ln2_g, const float* __restrict__ ln2_b,
    const float* __restrict__ b1,    const float* __restrict__ b2,
    const float* __restrict__ tln_g, const float* __restrict__ tln_b,
    const float* __restrict__ cls_w, const float* __restrict__ cls_b,
    const int* __restrict__ use_spmask_p, const int* __restrict__ grid_p,
    const _Float16* __restrict__ ws,
    float* __restrict__ out) {
  const int bimg = blockIdx.x;
  const int tid  = threadIdx.x;
  const int lane = tid & 31;
  const int wv   = tid >> 5;

  extern __shared__ char smem[];
  float*    tok = (float*)(smem + LDS_TOK);        // 208 x 64 f32
  _Float16* a16 = (_Float16*)(smem + LDS_A16);     // 208 x 64 f16
  char*     big = smem + LDS_BIG;

  // ===== Phase 1: image -> im2col -> conv(9x9,s2,p4)+GELU via WMMA =====
  float*    xs  = (float*)big;                     // 784 f32
  _Float16* im2 = (_Float16*)(big + 3200);         // 208 x 96 f16 (zero padded)
  _Float16* h16 = (_Float16*)(big + 43136);        // 208 x 160 f16

  for (int i = tid; i < 784; i += 256) xs[i] = x[bimg * 784 + i];
  __syncthreads();
  for (int i = tid; i < NPAD * 96; i += 256) {
    int p = i / 96, k = i - p * 96;
    float v = 0.f;
    if (p < NTOK && k < 81) {
      int oy = p / 14, ox = p - oy * 14;
      int ky = k / 9,  kx = k - ky * 9;
      int iy = oy * 2 - 4 + ky, ix = ox * 2 - 4 + kx;
      if (iy >= 0 && iy < 28 && ix >= 0 && ix < 28) v = xs[iy * 28 + ix];
    }
    im2[i] = (_Float16)v;
  }
  __syncthreads();

  // conv: 13 M-tiles x 10 N-tiles, two N-tiles per job (A reuse)
  {
    const _Float16* PW = ws + WS_CONVW;
    for (int job = wv; job < 13 * 5; job += 8) {
      int mt = job / 5, nt0 = (job - mt * 5) * 2;
      v8f acc0 = {}, acc1 = {};
#pragma unroll
      for (int kc = 0; kc < 3; ++kc) {
        v16h a  = load_a_frag<true>(im2, 96, mt * 16, kc * 32, lane);
        acc0 = wmma_f16(a, load_wfrag(PW, kc, 10, nt0,     lane), acc0);
        acc1 = wmma_f16(a, load_wfrag(PW, kc, 10, nt0 + 1, lane), acc1);
      }
      int n = nt0 * 16 + (lane & 15), hi = lane >> 4;
#pragma unroll
      for (int r = 0; r < 8; ++r) {
        int m = mt * 16 + r + (hi ? 8 : 0);
        h16[m * 160 + n]      = (_Float16)gelu_exact(acc0[r]);
        h16[m * 160 + n + 16] = (_Float16)gelu_exact(acc1[r]);
      }
    }
  }
  __syncthreads();

  // ===== Phase 2: 1x1 proj 150->64 -> tokens f32 =====
  {
    const _Float16* PW = ws + WS_PROJW;
    for (int job = wv; job < 13 * 4; job += 8) {
      int mt = job >> 2, nt = job & 3;
      v8f acc = {};
#pragma unroll
      for (int kc = 0; kc < 5; ++kc) {
        v16h a  = load_a_frag<true>(h16, 160, mt * 16, kc * 32, lane);
        acc = wmma_f16(a, load_wfrag(PW, kc, 4, nt, lane), acc);
      }
      int n = nt * 16 + (lane & 15), hi = lane >> 4;
#pragma unroll
      for (int r = 0; r < 8; ++r) {
        int m = mt * 16 + r + (hi ? 8 : 0);
        tok[m * DMODEL + n] = acc[r] + proj_b[n];
      }
    }
  }
  __syncthreads();

  // ===== Phase 3: LN1 -> a16, QKV GEMM (V stored transposed) =====
  ln_rows_to_f16(tok, a16, ln1_g, ln1_b, tid);
  __syncthreads();

  _Float16* qk = (_Float16*)big;               // 208 x 128 (Q cols 0..63 | K 64..127)
  _Float16* Vt = (_Float16*)(big + 53248);     // 64 x 224 (feature-major V)
  {
    const _Float16* PW = ws + WS_WQKV;
    for (int job = wv; job < 13 * 6; job += 8) {
      int mt = job / 6, nt0 = (job - mt * 6) * 2;
      v8f acc0 = {}, acc1 = {};
#pragma unroll
      for (int kc = 0; kc < 2; ++kc) {
        v16h a  = load_a_frag<true>(a16, DMODEL, mt * 16, kc * 32, lane);
        acc0 = wmma_f16(a, load_wfrag(PW, kc, 12, nt0,     lane), acc0);
        acc1 = wmma_f16(a, load_wfrag(PW, kc, 12, nt0 + 1, lane), acc1);
      }
      int n = nt0 * 16 + (lane & 15), hi = lane >> 4;
#pragma unroll
      for (int r = 0; r < 8; ++r) {
        int m = mt * 16 + r + (hi ? 8 : 0);
        _Float16 v0 = (_Float16)(acc0[r] + bqkv[n]);
        _Float16 v1 = (_Float16)(acc1[r] + bqkv[n + 16]);
        if (n < 128)      qk[m * 128 + n] = v0;
        else              Vt[(n - 128) * 224 + m] = v0;
        if (n + 16 < 128) qk[m * 128 + n + 16] = v1;
        else              Vt[(n - 112) * 224 + m] = v1;
      }
    }
  }
  __syncthreads();
  // zero Vt token columns 196..223 (pad for unconditional O-GEMM fragments)
  for (int i = tid; i < 64 * 28; i += 256) {
    int f = i / 28, t = 196 + (i - (i / 28) * 28);
    Vt[f * 224 + t] = (_Float16)0.0f;
  }
  __syncthreads();

  // ===== Phase 4: attention, per-wave (head,row-tile) jobs =====
  {
    char* wbuf = big + 81920 + wv * 20480;
    float*    sc = (float*)wbuf;               // 16 x 208 f32 scores
    _Float16* pr = (_Float16*)(wbuf + 13312);  // 16 x 224 f16 probs (zero padded)

    for (int job = wv; job < 52; job += 8) {
      int head = job & 3, mt = job >> 2;
      // scores tile = Q_tile (16 x 16, zero-padded K-chunk) x K^T
      v16h aq = load_a_frag<false>(qk, 128, mt * 16, head * 16, lane);
      for (int nt = 0; nt < 13; ++nt) {
        v8f acc = {};
        v16h bb = load_b_frag<false>(qk, 128, 64 + head * 16, nt * 16, lane);
        acc = wmma_f16(aq, bb, acc);
        int n = lane & 15, hi = lane >> 4;
#pragma unroll
        for (int r = 0; r < 8; ++r)
          sc[(r + (hi ? 8 : 0)) * NPAD + nt * 16 + n] = acc[r];
      }
      asm volatile("s_wait_dscnt 0" ::: "memory");
      // row softmax (scale 1/sqrt(16)=0.25), lanes 0..15 each own a row
      if (lane < 16) {
        float* row = sc + lane * NPAD;
        float mx = -1e30f;
        for (int c = 0; c < NTOK; ++c) mx = fmaxf(mx, row[c]);
        float mxs = mx * 0.25f;
        _Float16* prow = pr + lane * 224;
        float sum = 0.f;
        for (int c = 0; c < NTOK; ++c) {
          float e = __expf(row[c] * 0.25f - mxs);
          sum += e;
          prow[c] = (_Float16)e;
        }
        for (int c = NTOK; c < 224; ++c) prow[c] = (_Float16)0.0f;
        float inv = __builtin_amdgcn_rcpf(sum);
        for (int c = 0; c < NTOK; ++c)
          prow[c] = (_Float16)((float)prow[c] * inv);
      }
      asm volatile("s_wait_dscnt 0" ::: "memory");
      // O tile = probs (16 x 224) x V_head (224 x 16) -> a16[:, head*16:+16]
      v8f acc = {};
#pragma unroll
      for (int kc = 0; kc < 7; ++kc) {
        v16h a  = load_a_frag<true>(pr, 224, 0, kc * 32, lane);
        v16h bb = load_b_frag<true>(Vt, 224, kc * 32, head * 16, lane);
        acc = wmma_f16(a, bb, acc);
      }
      int n = lane & 15, hi = lane >> 4;
#pragma unroll
      for (int r = 0; r < 8; ++r) {
        int m = mt * 16 + r + (hi ? 8 : 0);
        a16[m * DMODEL + head * 16 + n] = (_Float16)acc[r];
      }
    }
  }
  __syncthreads();

  // ===== Phase 5: tokens += O @ Wo^T + bo =====
  {
    const _Float16* PW = ws + WS_WO;
    for (int job = wv; job < 13 * 4; job += 8) {
      int mt = job >> 2, nt = job & 3;
      v8f acc = {};
#pragma unroll
      for (int kc = 0; kc < 2; ++kc) {
        v16h a  = load_a_frag<true>(a16, DMODEL, mt * 16, kc * 32, lane);
        acc = wmma_f16(a, load_wfrag(PW, kc, 4, nt, lane), acc);
      }
      int n = nt * 16 + (lane & 15), hi = lane >> 4;
#pragma unroll
      for (int r = 0; r < 8; ++r) {
        int m = mt * 16 + r + (hi ? 8 : 0);
        tok[m * DMODEL + n] += acc[r] + bo[n];
      }
    }
  }
  __syncthreads();

  // ===== Phase 6: LN2 -> a16, MLP 64->256 (GELU) -> 64, residual =====
  ln_rows_to_f16(tok, a16, ln2_g, ln2_b, tid);
  __syncthreads();

  _Float16* hid16 = (_Float16*)big;            // 208 x 256 f16
  {
    const _Float16* PW = ws + WS_W1;
    for (int job = wv; job < 13 * 8; job += 8) {
      int mt = job >> 3, nt0 = (job & 7) * 2;
      v8f acc0 = {}, acc1 = {};
#pragma unroll
      for (int kc = 0; kc < 2; ++kc) {
        v16h a  = load_a_frag<true>(a16, DMODEL, mt * 16, kc * 32, lane);
        acc0 = wmma_f16(a, load_wfrag(PW, kc, 16, nt0,     lane), acc0);
        acc1 = wmma_f16(a, load_wfrag(PW, kc, 16, nt0 + 1, lane), acc1);
      }
      int n = nt0 * 16 + (lane & 15), hi = lane >> 4;
#pragma unroll
      for (int r = 0; r < 8; ++r) {
        int m = mt * 16 + r + (hi ? 8 : 0);
        hid16[m * 256 + n]      = (_Float16)gelu_exact(acc0[r] + b1[n]);
        hid16[m * 256 + n + 16] = (_Float16)gelu_exact(acc1[r] + b1[n + 16]);
      }
    }
  }
  __syncthreads();
  {
    const _Float16* PW = ws + WS_W2;
    for (int job = wv; job < 13 * 4; job += 8) {
      int mt = job >> 2, nt = job & 3;
      v8f acc = {};
#pragma unroll
      for (int kc = 0; kc < 8; ++kc) {
        v16h a  = load_a_frag<true>(hid16, 256, mt * 16, kc * 32, lane);
        acc = wmma_f16(a, load_wfrag(PW, kc, 4, nt, lane), acc);
      }
      int n = nt * 16 + (lane & 15), hi = lane >> 4;
#pragma unroll
      for (int r = 0; r < 8; ++r) {
        int m = mt * 16 + r + (hi ? 8 : 0);
        tok[m * DMODEL + n] += acc[r] + b2[n];
      }
    }
  }
  __syncthreads();

  // ===== Phase 7: final LN (in place, f32) + l2n copy (f16) in a16 =====
  for (int row = tid; row < NPAD; row += 256) {
    _Float16* o = a16 + row * DMODEL;
    if (row < NTOK) {
      float* t = tok + row * DMODEL;
      float mu = 0.f;
      for (int c = 0; c < DMODEL; ++c) mu += t[c];
      mu *= (1.0f / DMODEL);
      float var = 0.f;
      for (int c = 0; c < DMODEL; ++c) { float d = t[c] - mu; var += d * d; }
      var *= (1.0f / DMODEL);
      float r = rsqrtf(var + 1e-5f);
      float n2 = 0.f;
      for (int c = 0; c < DMODEL; ++c) {
        float y = (t[c] - mu) * r * tln_g[c] + tln_b[c];
        t[c] = y; n2 += y * y;
      }
      float inv = 1.0f / fmaxf(sqrtf(n2), 1e-12f);
      for (int c = 0; c < DMODEL; ++c) o[c] = (_Float16)(t[c] * inv);
    } else {
      for (int c = 0; c < DMODEL; ++c) o[c] = (_Float16)0.0f;
    }
  }
  __syncthreads();

  // ===== Phase 8: slot logits = l2n(slot_q) @ l2n(tokens)^T / 8 =====
  float* alog = (float*)big;                   // 16 x 208 f32
  {
    const _Float16* PA = ws + WS_SLOTA;
    for (int nt = wv; nt < 13; nt += 8) {
      v8f acc = {};
#pragma unroll
      for (int kc = 0; kc < 2; ++kc) {
        v16h a  = *(const v16h*)(PA + (kc * 32 + lane) * 16);
        v16h bb = load_b_frag<true>(a16, DMODEL, kc * 32, nt * 16, lane);
        acc = wmma_f16(a, bb, acc);
      }
      int n = nt * 16 + (lane & 15), hi = lane >> 4;
#pragma unroll
      for (int r = 0; r < 8; ++r)
        alog[(r + (hi ? 8 : 0)) * NPAD + n] = acc[r] * 0.125f;
    }
  }
  __syncthreads();

  // ===== Phase 9: slot softmax, spatial mask, P, S, classification =====
  float* sbuf = (float*)(big + 13312);         // 12 x 64 f32
  float* zbuf = (float*)(big + 16384);         // 12 f32
  float* pbuf = (float*)(big + 16448);         // 12 f32
  const int use_sp = use_spmask_p[0];
  int g = grid_p[0]; if (g < 1) g = 1; if (g > 4) g = 4;

  if (tid < 12) {
    float* row = alog + tid * NPAD;
    float mx = -1e30f;
    for (int n = 0; n < NTOK; ++n) mx = fmaxf(mx, row[n]);
    float sum = 0.f;
    for (int n = 0; n < NTOK; ++n) {
      float e = __expf(row[n] - mx);
      row[n] = e; sum += e;
    }
    float inv = __builtin_amdgcn_rcpf(sum);
    for (int n = 0; n < NTOK; ++n) row[n] *= inv;   // A_slot
    float z;
    if (use_sp) {
      int xb[5];
      for (int c = 0; c <= g; ++c) xb[c] = (int)roundf(c * (14.0f / g));
      const int G = g * g;
      float mass[16];
      for (int i = 0; i < G; ++i) mass[i] = 0.f;
      for (int n = 0; n < NTOK; ++n) {
        int y = n / 14, xq = n - 14 * y;
        int cy = 0; while (cy + 1 < g && y  >= xb[cy + 1]) ++cy;
        int cx = 0; while (cx + 1 < g && xq >= xb[cx + 1]) ++cx;
        mass[cy * g + cx] += row[n];
      }
      int i1 = 0;
      for (int i = 1; i < G; ++i) if (mass[i] > mass[i1]) i1 = i;
      int i2 = (i1 == 0) ? 1 : 0;
      for (int i = 0; i < G; ++i) if (i != i1 && mass[i] > mass[i2]) i2 = i;
      float s = 0.f;
      for (int n = 0; n < NTOK; ++n) {
        int y = n / 14, xq = n - 14 * y;
        int cy = 0; while (cy + 1 < g && y  >= xb[cy + 1]) ++cy;
        int cx = 0; while (cx + 1 < g && xq >= xb[cx + 1]) ++cx;
        int cell = cy * g + cx;
        float a = (cell == i1 || cell == i2) ? row[n] : 0.f;
        row[n] = a; s += a;
      }
      float sc2 = fmaxf(s, 1e-8f);
      float rs = 1.0f / sc2;
      float m2 = 0.f;
      for (int n = 0; n < NTOK; ++n) { float a = row[n] * rs; row[n] = a; m2 += a; }
      z = m2;
    } else {
      z = mx + __logf(sum);                     // logsumexp of logits
    }
    zbuf[tid] = z;
  }
  __syncthreads();
  if (tid == 0) {
    float mean = 0.f;
    for (int m = 0; m < 12; ++m) mean += zbuf[m];
    mean *= (1.0f / 12.0f);
    float zz[12]; float mx = -1e30f;
    for (int m = 0; m < 12; ++m) { zz[m] = (zbuf[m] - mean) * 2.0f; mx = fmaxf(mx, zz[m]); }
    float sum = 0.f;
    for (int m = 0; m < 12; ++m) { zz[m] = __expf(zz[m] - mx); sum += zz[m]; }
    float inv = __builtin_amdgcn_rcpf(sum);
    for (int m = 0; m < 12; ++m) pbuf[m] = zz[m] * inv;
  }
  __syncthreads();
  // S = A_eff @ tokens  (12 x 64)
  for (int o = tid; o < 12 * DMODEL; o += 256) {
    int m = o >> 6, c = o & 63;
    const float* ar = alog + m * NPAD;
    float s = 0.f;
    for (int n = 0; n < NTOK; ++n) s += ar[n] * tok[n * DMODEL + c];
    sbuf[o] = s;
  }
  __syncthreads();
  if (tid < 12) {
    float n2 = 0.f;
    for (int c = 0; c < DMODEL; ++c) { float v = sbuf[tid * DMODEL + c]; n2 += v * v; }
    float inv = 1.0f / fmaxf(sqrtf(n2), 1e-12f);
    for (int c = 0; c < DMODEL; ++c) sbuf[tid * DMODEL + c] *= inv;
  }
  __syncthreads();
  if (tid < 47) {
    float acc = 0.f;
    const float* wrow = cls_w + tid * DMODEL;
    for (int m = 0; m < 12; ++m) {
      const float* srow = sbuf + m * DMODEL;
      float d = cls_b[tid];
      for (int c = 0; c < DMODEL; ++c) d += srow[c] * wrow[c];
      acc += pbuf[m] * d;
    }
    out[bimg * 47 + tid] = acc;
  }
}

// ---------------------------------------------------------------------------
extern "C" void kernel_launch(void* const* d_in, const int* in_sizes, int n_in,
                              void* d_out, int out_size, void* d_ws, size_t ws_size,
                              hipStream_t stream) {
  const float* x      = (const float*)d_in[0];
  const float* conv1w = (const float*)d_in[1];
  const float* projw  = (const float*)d_in[2];
  const float* projb  = (const float*)d_in[3];
  const float* ln1g   = (const float*)d_in[4];
  const float* ln1b   = (const float*)d_in[5];
  const float* wqkv   = (const float*)d_in[6];
  const float* bqkv   = (const float*)d_in[7];
  const float* wo     = (const float*)d_in[8];
  const float* bo     = (const float*)d_in[9];
  const float* ln2g   = (const float*)d_in[10];
  const float* ln2b   = (const float*)d_in[11];
  const float* w1     = (const float*)d_in[12];
  const float* b1     = (const float*)d_in[13];
  const float* w2     = (const float*)d_in[14];
  const float* b2     = (const float*)d_in[15];
  const float* tlng   = (const float*)d_in[16];
  const float* tlnb   = (const float*)d_in[17];
  const float* slotq  = (const float*)d_in[18];
  const float* clsw   = (const float*)d_in[19];
  const float* clsb   = (const float*)d_in[20];
  const int*   usesp  = (const int*)d_in[21];
  const int*   grid   = (const int*)d_in[22];

  _Float16* wsh = (_Float16*)d_ws;
  const int B = in_sizes[0] / 784;

  slotnet_prep<<<1, 256, 0, stream>>>(conv1w, projw, wqkv, wo, w1, w2, slotq, wsh);

  (void)hipFuncSetAttribute((const void*)slotnet_main,
                            hipFuncAttributeMaxDynamicSharedMemorySize, LDS_TOTAL);
  slotnet_main<<<B, 256, LDS_TOTAL, stream>>>(
      x, projb, ln1g, ln1b, bqkv, bo, ln2g, ln2b, b1, b2, tlng, tlnb,
      clsw, clsb, usesp, grid, wsh, (float*)d_out);

  (void)n_in; (void)out_size; (void)ws_size;
}
